// SpatioTemporalTransformer_44659069944066
// MI455X (gfx1250) — compile-verified, matrix operand order checked by reading, not run on hardware
//
#include <hip/hip_runtime.h>
#include <hip/hip_bf16.h>
#include <math.h>

// ---------------- problem constants ----------------
#define B_   2
#define T_   16
#define N_   2000
#define FIN_ 16
#define E_   32000
#define H_   4
#define C_   32
#define D_   64
#define NH_  4
#define FF_  2048
#define G_   (B_ * T_)     // 32 graph replicas
#define GN_  (G_ * N_)     // 64000 rows
#define ET_  (E_ + N_)     // 34000 edges incl self loops
#define BN_  (B_ * N_)     // 4000 sequences

typedef __attribute__((ext_vector_type(16))) __bf16 bf16x16;
typedef __attribute__((ext_vector_type(8)))  __bf16 bf16x8;
typedef __attribute__((ext_vector_type(8)))  float  f32x8;

// ---------------- WMMA fragment helpers (CDNA5 16-bit layouts) ----------------
__device__ __forceinline__ bf16x16 pack_a(float4 p0, float4 p1, float4 p2, float4 p3) {
  bf16x16 a;
  a[0]  = (__bf16)p0.x; a[1]  = (__bf16)p0.y; a[2]  = (__bf16)p0.z; a[3]  = (__bf16)p0.w;
  a[4]  = (__bf16)p1.x; a[5]  = (__bf16)p1.y; a[6]  = (__bf16)p1.z; a[7]  = (__bf16)p1.w;
  a[8]  = (__bf16)p2.x; a[9]  = (__bf16)p2.y; a[10] = (__bf16)p2.z; a[11] = (__bf16)p2.w;
  a[12] = (__bf16)p3.x; a[13] = (__bf16)p3.y; a[14] = (__bf16)p3.z; a[15] = (__bf16)p3.w;
  return a;
}

// A 16x32 fragment: per lane two contiguous 8-float runs, loaded as 4x b128.
// Only uniform (scalar) branch for the K=16 tail case.
__device__ __forceinline__ bf16x16 load_afrag(const float* __restrict__ arow, int k0, int Kreal) {
  float4 p0 = *(const float4*)(arow + k0);
  float4 p1 = *(const float4*)(arow + k0 + 4);
  float4 p2, p3;
  if (k0 + 16 < Kreal) {
    p2 = *(const float4*)(arow + k0 + 16);
    p3 = *(const float4*)(arow + k0 + 20);
  } else {
    p2 = make_float4(0.f, 0.f, 0.f, 0.f);
    p3 = p2;
  }
  return pack_a(p0, p1, p2, p3);
}

// B fragment from pre-transposed bf16 weights Bt[N][Kpad]: two 16B loads.
__device__ __forceinline__ bf16x16 load_bfrag(const __bf16* __restrict__ bp, int k0) {
  bf16x8 q0 = *(const bf16x8*)(bp + k0);
  bf16x8 q1 = *(const bf16x8*)(bp + k0 + 16);
  return __builtin_shufflevector(q0, q1, 0, 1, 2, 3, 4, 5, 6, 7, 8, 9, 10, 11, 12, 13, 14, 15);
}

// weight conversion: Bt[n*Kpad + k] = bf16(W[k*ldw + colB0 + n]), zero-padded K
__global__ void k_convw(const float* __restrict__ W, int ldw, int colB0,
                        int Ncols, int Kreal, int Kpad, __bf16* __restrict__ Bt) {
  int idx = blockIdx.x * blockDim.x + threadIdx.x;
  if (idx >= Ncols * Kpad) return;
  int n = idx / Kpad, k = idx - n * Kpad;
  float v = (k < Kreal) ? W[(size_t)k * ldw + colB0 + n] : 0.0f;
  Bt[idx] = (__bf16)v;
}

// ---------------- bf16 WMMA GEMM, 16x32 tile per wave ----------------
// C = A(MxK) * B(KxN) [+bias][+resid][relu]; B pre-transposed bf16 [N][Kpad]
__global__ void k_gemm_bf16(const float* __restrict__ A, int lda,
                            const __bf16* __restrict__ Bt, int Kpad,
                            const float* __restrict__ bias, int bias0,
                            const float* __restrict__ resid,
                            float* __restrict__ Cm, int ldc,
                            int M, int Ncols, int Kreal, int relu) {
  int wave = blockIdx.x * (blockDim.x >> 5) + (threadIdx.x >> 5);
  int tilesN = Ncols >> 5;
  int tiles = (M >> 4) * tilesN;
  if (wave >= tiles) return;
  int row0 = (wave / tilesN) << 4;
  int col0 = (wave % tilesN) << 5;
  int lane = threadIdx.x & 31;
  int half = (lane >> 4) & 1;
  const float*  arow = A  + (size_t)(row0 + (lane & 15)) * lda + half * 8;
  const __bf16* b0p  = Bt + (size_t)(col0 + (lane & 15)) * Kpad + half * 8;
  const __bf16* b1p  = b0p + (size_t)16 * Kpad;
  f32x8 acc0 = {}, acc1 = {};
  for (int k0 = 0; k0 < Kpad; k0 += 32) {
    __builtin_prefetch(arow + k0 + 32, 0, 1);
    bf16x16 a  = load_afrag(arow, k0, Kreal);
    bf16x16 f0 = load_bfrag(b0p, k0);
    bf16x16 f1 = load_bfrag(b1p, k0);
    acc0 = __builtin_amdgcn_wmma_f32_16x16x32_bf16(false, a, false, f0, (short)0, acc0, false, false);
    acc1 = __builtin_amdgcn_wmma_f32_16x16x32_bf16(false, a, false, f1, (short)0, acc1, false, false);
  }
  int colA = col0 + (lane & 15);
  int colB = colA + 16;
  int rbase = row0 + (half ? 8 : 0);
#pragma unroll
  for (int r = 0; r < 8; ++r) {
    int row = rbase + r;
    float v0 = acc0[r], v1 = acc1[r];
    if (bias)  { v0 += bias[bias0 + colA]; v1 += bias[bias0 + colB]; }
    if (resid) { v0 += resid[(size_t)row * ldc + colA]; v1 += resid[(size_t)row * ldc + colB]; }
    if (relu)  { v0 = fmaxf(v0, 0.0f); v1 = fmaxf(v1, 0.0f); }
    Cm[(size_t)row * ldc + colA] = v0;
    Cm[(size_t)row * ldc + colB] = v1;
  }
}

// ---------------- projection GEMM with permuted store ----------------
// [g*N+n,32] x [32,64] -> y[((b*N+n)*T + t)*64]
__global__ void k_proj_gemm(const float* __restrict__ A, const __bf16* __restrict__ Wpt,
                            const float* __restrict__ bp, float* __restrict__ Y) {
  int wave = blockIdx.x * (blockDim.x >> 5) + (threadIdx.x >> 5);
  const int tilesN = D_ / 32;  // 2
  int tiles = (GN_ / 16) * tilesN;
  if (wave >= tiles) return;
  int row0 = (wave / tilesN) << 4;
  int col0 = (wave % tilesN) << 5;
  int lane = threadIdx.x & 31;
  int half = (lane >> 4) & 1;
  const float*  arow = A + (size_t)(row0 + (lane & 15)) * C_ + half * 8;
  const __bf16* b0p  = Wpt + (size_t)(col0 + (lane & 15)) * C_ + half * 8;
  const __bf16* b1p  = b0p + (size_t)16 * C_;
  f32x8 acc0 = {}, acc1 = {};
  bf16x16 a  = load_afrag(arow, 0, C_);
  bf16x16 f0 = load_bfrag(b0p, 0);
  bf16x16 f1 = load_bfrag(b1p, 0);
  acc0 = __builtin_amdgcn_wmma_f32_16x16x32_bf16(false, a, false, f0, (short)0, acc0, false, false);
  acc1 = __builtin_amdgcn_wmma_f32_16x16x32_bf16(false, a, false, f1, (short)0, acc1, false, false);
  int colA = col0 + (lane & 15);
  int colB = colA + 16;
  int rbase = row0 + (half ? 8 : 0);
#pragma unroll
  for (int r = 0; r < 8; ++r) {
    int row = rbase + r;                      // row = g*N + n (N%16==0 so tile shares g)
    int g = row / N_, n = row - g * N_;
    int bb = g / T_, t = g - bb * T_;
    size_t base = (((size_t)(bb * N_ + n)) * T_ + t) * D_;
    Y[base + colA] = acc0[r] + bp[colA];
    Y[base + colB] = acc1[r] + bp[colB];
  }
}

// ---------------- scalar helper math ----------------
__device__ __forceinline__ unsigned ford(float f) {
  unsigned u = __float_as_uint(f);
  return (u & 0x80000000u) ? ~u : (u | 0x80000000u);
}
__device__ __forceinline__ float ordf(unsigned e) {
  unsigned u = (e & 0x80000000u) ? (e & 0x7fffffffu) : ~e;
  return __uint_as_float(u);
}

// ---------------- init ----------------
__global__ void k_init(float* __restrict__ out_gat, const float* __restrict__ gat_bias,
                       float* __restrict__ den, unsigned* __restrict__ mbuf,
                       float* __restrict__ ewsum) {
  int idx = blockIdx.x * blockDim.x + threadIdx.x;
  if (idx < GN_ * C_) out_gat[idx] = gat_bias[idx & (C_ - 1)];
  if (idx < GN_ * H_) { den[idx] = 0.0f; mbuf[idx] = 0x007FFFFFu; }  // encode(-inf)
  if (idx == 0) ewsum[0] = 0.0f;
}

__global__ void k_ewsum(const float* __restrict__ ew, float* __restrict__ ewsum) {
  int e = blockIdx.x * blockDim.x + threadIdx.x;
  if (e < E_) atomicAdd(ewsum, ew[e]);
}

__global__ void k_edgeprep(const int* __restrict__ ei, const float* __restrict__ ew,
                           const float* __restrict__ W_edge, const float* __restrict__ att_edge,
                           const float* __restrict__ ewsum,
                           int* __restrict__ sbuf, int* __restrict__ dbuf,
                           float* __restrict__ aedge) {
  int e = blockIdx.x * blockDim.x + threadIdx.x;
  if (e >= ET_) return;
  int s, d2; float w;
  if (e < E_) { s = ei[e]; d2 = ei[E_ + e]; w = ew[e]; }
  else        { s = d2 = e - E_; w = ewsum[0] * (1.0f / E_); }
  sbuf[e] = s; dbuf[e] = d2;
#pragma unroll
  for (int h = 0; h < H_; ++h) {
    float c = 0.0f;
    for (int j = 0; j < C_; ++j) c += W_edge[h * C_ + j] * att_edge[h * C_ + j];
    aedge[e * H_ + h] = w * c;
  }
}

__global__ void k_attcoef(const float* __restrict__ hbuf, const float* __restrict__ att_src,
                          const float* __restrict__ att_dst,
                          float* __restrict__ asrc, float* __restrict__ adst) {
  int idx = blockIdx.x * blockDim.x + threadIdx.x;
  if (idx >= GN_ * H_) return;
  int h = idx & (H_ - 1);
  int gn = idx >> 2;
  const float* hp = hbuf + (size_t)gn * (H_ * C_) + h * C_;
  float s1 = 0.0f, s2 = 0.0f;
  for (int c = 0; c < C_; ++c) { s1 += hp[c] * att_src[h * C_ + c]; s2 += hp[c] * att_dst[h * C_ + c]; }
  asrc[idx] = s1; adst[idx] = s2;
}

__global__ void k_escore(const int* __restrict__ sbuf, const int* __restrict__ dbuf,
                         const float* __restrict__ asrc, const float* __restrict__ adst,
                         const float* __restrict__ aedge,
                         float* __restrict__ ebuf, unsigned* __restrict__ mbuf) {
  int idx = blockIdx.x * blockDim.x + threadIdx.x;
  if (idx >= ET_ * G_) return;
  int e = idx % ET_, g = idx / ET_;
  int s = sbuf[e], d2 = dbuf[e];
#pragma unroll
  for (int h = 0; h < H_; ++h) {
    float v = asrc[(g * N_ + s) * H_ + h] + adst[(g * N_ + d2) * H_ + h] + aedge[e * H_ + h];
    v = (v > 0.0f) ? v : 0.2f * v;
    ebuf[(size_t)idx * H_ + h] = v;
    atomicMax(&mbuf[(g * N_ + d2) * H_ + h], ford(v));
  }
}

__global__ void k_expsum(const int* __restrict__ dbuf, float* __restrict__ ebuf,
                         const unsigned* __restrict__ mbuf, float* __restrict__ den) {
  int idx = blockIdx.x * blockDim.x + threadIdx.x;
  if (idx >= ET_ * G_) return;
  int e = idx % ET_, g = idx / ET_;
  int d2 = dbuf[e];
#pragma unroll
  for (int h = 0; h < H_; ++h) {
    float m = ordf(mbuf[(g * N_ + d2) * H_ + h]);
    float ex = __expf(ebuf[(size_t)idx * H_ + h] - m);
    ebuf[(size_t)idx * H_ + h] = ex;
    atomicAdd(&den[(g * N_ + d2) * H_ + h], ex);
  }
}

__global__ void k_agg(const int* __restrict__ sbuf, const int* __restrict__ dbuf,
                      const float* __restrict__ ebuf, const float* __restrict__ den,
                      const float* __restrict__ hbuf, float* __restrict__ out_gat) {
  int idx = blockIdx.x * blockDim.x + threadIdx.x;
  if (idx >= ET_ * G_) return;
  int e = idx % ET_, g = idx / ET_;
  int s = sbuf[e], d2 = dbuf[e];
  float at[H_];
#pragma unroll
  for (int h = 0; h < H_; ++h)
    at[h] = ebuf[(size_t)idx * H_ + h] / (den[(g * N_ + d2) * H_ + h] + 1e-16f) * 0.25f;
  const float* hp = hbuf + (size_t)(g * N_ + s) * (H_ * C_);
  float* op = out_gat + (size_t)(g * N_ + d2) * C_;
  for (int c = 0; c < C_; ++c) {
    float a = hp[c] * at[0] + hp[C_ + c] * at[1] + hp[2 * C_ + c] * at[2] + hp[3 * C_ + c] * at[3];
    atomicAdd(&op[c], a);
  }
}

// last-timestep attention: one thread per (sequence, head)
__global__ void k_attn(const float* __restrict__ y_perm, const float* __restrict__ Wqkv,
                       const float* __restrict__ bqkv, const float* __restrict__ kv,
                       float* __restrict__ ctx, float* __restrict__ ylast) {
  int idx = blockIdx.x * blockDim.x + threadIdx.x;
  if (idx >= BN_ * NH_) return;
  int bn = idx / NH_, hd = idx % NH_;
  const int dh = D_ / NH_;  // 16
  const float* yl = y_perm + ((size_t)bn * T_ + (T_ - 1)) * D_;
  if (hd == 0)
    for (int d2 = 0; d2 < D_; ++d2) ylast[(size_t)bn * D_ + d2] = yl[d2];
  float q[dh];
  for (int j = 0; j < dh; ++j) {
    int col = hd * dh + j;
    float s = bqkv[col];
    for (int d2 = 0; d2 < D_; ++d2) s += yl[d2] * Wqkv[d2 * (3 * D_) + col];
    q[j] = s;
  }
  float sc[T_]; float mx = -1e30f;
  for (int t = 0; t < T_; ++t) {
    const float* kp = kv + ((size_t)bn * T_ + t) * 128 + hd * dh;
    float s = 0.0f;
    for (int j = 0; j < dh; ++j) s += q[j] * kp[j];
    s *= 0.25f;  // 1/sqrt(16)
    sc[t] = s; mx = fmaxf(mx, s);
  }
  float dsum = 0.0f;
  for (int t = 0; t < T_; ++t) { sc[t] = __expf(sc[t] - mx); dsum += sc[t]; }
  float inv = 1.0f / dsum;
  for (int j = 0; j < dh; ++j) {
    float s = 0.0f;
    for (int t = 0; t < T_; ++t) s += sc[t] * kv[((size_t)bn * T_ + t) * 128 + 64 + hd * dh + j];
    ctx[(size_t)bn * D_ + hd * dh + j] = s * inv;
  }
}

// layernorm over D=64 (biased variance, eps 1e-5), optional residual
__global__ void k_ln(const float* __restrict__ x, const float* __restrict__ resid,
                     const float* __restrict__ gam, const float* __restrict__ bet,
                     float* __restrict__ out, int rows) {
  int r = blockIdx.x * blockDim.x + threadIdx.x;
  if (r >= rows) return;
  float buf[D_]; float mu = 0.0f;
  for (int d = 0; d < D_; ++d) {
    float v = x[(size_t)r * D_ + d] + (resid ? resid[(size_t)r * D_ + d] : 0.0f);
    buf[d] = v; mu += v;
  }
  mu *= (1.0f / D_);
  float var = 0.0f;
  for (int d = 0; d < D_; ++d) { float dv = buf[d] - mu; var += dv * dv; }
  var *= (1.0f / D_);
  float inv = rsqrtf(var + 1e-5f);
  for (int d = 0; d < D_; ++d)
    out[(size_t)r * D_ + d] = (buf[d] - mu) * inv * gam[d] + bet[d];
}

// ---------------- host side ----------------
static inline int cdiv(long long a, long long b) { return (int)((a + b - 1) / b); }

extern "C" void kernel_launch(void* const* d_in, const int* in_sizes, int n_in,
                              void* d_out, int out_size, void* d_ws, size_t ws_size,
                              hipStream_t stream) {
  (void)in_sizes; (void)n_in; (void)out_size; (void)ws_size;
  const float* x_seq    = (const float*)d_in[0];
  const int*   edge_idx = (const int*)  d_in[1];
  const float* edge_w   = (const float*)d_in[2];
  const float* W_gat    = (const float*)d_in[3];
  const float* att_src  = (const float*)d_in[4];
  const float* att_dst  = (const float*)d_in[5];
  const float* W_edge   = (const float*)d_in[6];
  const float* att_edge = (const float*)d_in[7];
  const float* gat_bias = (const float*)d_in[8];
  const float* W_proj   = (const float*)d_in[9];
  const float* b_proj   = (const float*)d_in[10];
  const float* Wqkv     = (const float*)d_in[11];
  const float* bqkv     = (const float*)d_in[12];
  const float* Wo       = (const float*)d_in[13];
  const float* bo       = (const float*)d_in[14];
  const float* ln1_g    = (const float*)d_in[15];
  const float* ln1_b    = (const float*)d_in[16];
  const float* W1       = (const float*)d_in[17];
  const float* b1       = (const float*)d_in[18];
  const float* W2       = (const float*)d_in[19];
  const float* b2       = (const float*)d_in[20];
  const float* ln2_g    = (const float*)d_in[21];
  const float* ln2_b    = (const float*)d_in[22];
  float* out = (float*)d_out;

  // workspace carve-up (float region, all offsets multiples of 4 -> 16B aligned)
  float* f = (float*)d_ws;
  size_t o = 0;
  float* hbuf    = f + o; o += (size_t)GN_ * 128;   // h features; reused later as kvbuf
  float* asrc    = f + o; o += (size_t)GN_ * H_;
  float* adst    = f + o; o += (size_t)GN_ * H_;
  float* aedge   = f + o; o += (size_t)ET_ * H_;
  float* den     = f + o; o += (size_t)GN_ * H_;
  float* out_gat = f + o; o += (size_t)GN_ * C_;
  float* y_perm  = f + o; o += (size_t)GN_ * D_;    // [(b*N+n)*T + t, D]
  float* ylast   = f + o; o += (size_t)BN_ * D_;
  float* ctx     = f + o; o += (size_t)BN_ * D_;
  float* tmp1    = f + o; o += (size_t)BN_ * D_;
  float* z1      = f + o; o += (size_t)BN_ * D_;
  float* act     = f + o; o += (size_t)BN_ * FF_;
  float* ffout   = f + o; o += (size_t)BN_ * D_;
  float* ebuf    = f + o; o += (size_t)ET_ * G_ * H_;
  float* ewsum   = f + o; o += 16;
  unsigned* mbuf = (unsigned*)(f + o); o += (size_t)GN_ * H_;
  int* sbuf      = (int*)(f + o); o += ET_;
  int* dbuf      = (int*)(f + o); o += ET_ + 2;     // +2: leaves next region only 8B (not 16B) aligned
  // bf16 transposed weight region
  __bf16* bw = (__bf16*)(f + o);
  size_t bo2 = 0;
  __bf16* Wg_t  = bw + bo2; bo2 += (size_t)128 * 32;    // [128][Kpad=32], Kreal=16
  __bf16* Wkv_t = bw + bo2; bo2 += (size_t)128 * 64;    // Wqkv cols 64..191
  __bf16* Wo_t  = bw + bo2; bo2 += (size_t)64 * 64;
  __bf16* W1_t  = bw + bo2; bo2 += (size_t)FF_ * 64;
  __bf16* W2_t  = bw + bo2; bo2 += (size_t)64 * FF_;
  __bf16* Wp_t  = bw + bo2; bo2 += (size_t)64 * 32;
  float* kvbuf = hbuf;  // reuse: h dead after k_agg, kv needed after

  const int TPB = 256;
  const int WPB = TPB / 32;

  // ---- weight conversions (independent; run first) ----
  k_convw<<<cdiv(128 * 32, TPB), TPB, 0, stream>>>(W_gat, 128, 0, 128, 16, 32, Wg_t);
  k_convw<<<cdiv(128 * 64, TPB), TPB, 0, stream>>>(Wqkv, 192, 64, 128, 64, 64, Wkv_t);
  k_convw<<<cdiv(64 * 64, TPB), TPB, 0, stream>>>(Wo, 64, 0, 64, 64, 64, Wo_t);
  k_convw<<<cdiv(FF_ * 64, TPB), TPB, 0, stream>>>(W1, FF_, 0, FF_, 64, 64, W1_t);
  k_convw<<<cdiv(64 * FF_, TPB), TPB, 0, stream>>>(W2, 64, 0, 64, FF_, FF_, W2_t);
  k_convw<<<cdiv(64 * 32, TPB), TPB, 0, stream>>>(W_proj, 64, 0, 64, 32, 32, Wp_t);

  // ---- init ----
  k_init<<<cdiv((long long)GN_ * C_, TPB), TPB, 0, stream>>>(out_gat, gat_bias, den, mbuf, ewsum);
  k_ewsum<<<cdiv(E_, TPB), TPB, 0, stream>>>(edge_w, ewsum);
  k_edgeprep<<<cdiv(ET_, TPB), TPB, 0, stream>>>(edge_idx, edge_w, W_edge, att_edge, ewsum,
                                                 sbuf, dbuf, aedge);

  // h = x @ W_gat  (K=16 zero-padded to 32)  [64000x16]x[16x128]
  {
    int tiles = (GN_ / 16) * (128 / 32);
    k_gemm_bf16<<<cdiv(tiles, WPB), TPB, 0, stream>>>(
        x_seq, FIN_, Wg_t, 32, nullptr, 0, nullptr, hbuf, 128, GN_, 128, FIN_, 0);
  }
  k_attcoef<<<cdiv((long long)GN_ * H_, TPB), TPB, 0, stream>>>(hbuf, att_src, att_dst, asrc, adst);

  // edge softmax + aggregation
  k_escore<<<cdiv((long long)ET_ * G_, TPB), TPB, 0, stream>>>(sbuf, dbuf, asrc, adst, aedge, ebuf, mbuf);
  k_expsum<<<cdiv((long long)ET_ * G_, TPB), TPB, 0, stream>>>(dbuf, ebuf, mbuf, den);
  k_agg<<<cdiv((long long)ET_ * G_, TPB), TPB, 0, stream>>>(sbuf, dbuf, ebuf, den, hbuf, out_gat);

  // y = out_gat @ W_proj + b_proj, stored permuted
  {
    int tiles = (GN_ / 16) * (D_ / 32);
    k_proj_gemm<<<cdiv(tiles, WPB), TPB, 0, stream>>>(out_gat, Wp_t, b_proj, y_perm);
  }

  // K/V for all positions: kv = y @ Wqkv[:,64:192] + bqkv[64:192]
  {
    int tiles = (GN_ / 16) * (128 / 32);
    k_gemm_bf16<<<cdiv(tiles, WPB), TPB, 0, stream>>>(
        y_perm, D_, Wkv_t, 64, bqkv, 64, nullptr, kvbuf, 128, GN_, 128, D_, 0);
  }

  // last-token attention -> ctx, and extract ylast
  k_attn<<<cdiv((long long)BN_ * NH_, TPB), TPB, 0, stream>>>(y_perm, Wqkv, bqkv, kvbuf, ctx, ylast);

  // tmp1 = ylast + ctx @ Wo + bo ; z1 = LN1(tmp1)
  {
    int tiles = (BN_ / 16) * (D_ / 32);
    k_gemm_bf16<<<cdiv(tiles, WPB), TPB, 0, stream>>>(
        ctx, D_, Wo_t, 64, bo, 0, ylast, tmp1, D_, BN_, D_, D_, 0);
  }
  k_ln<<<cdiv(BN_, TPB), TPB, 0, stream>>>(tmp1, nullptr, ln1_g, ln1_b, z1, BN_);

  // act = relu(z1 @ W1 + b1)   [4000x64]x[64x2048]
  {
    int tiles = (BN_ / 16) * (FF_ / 32);
    k_gemm_bf16<<<cdiv(tiles, WPB), TPB, 0, stream>>>(
        z1, D_, W1_t, 64, b1, 0, nullptr, act, FF_, BN_, FF_, D_, 1);
  }
  // ffout = act @ W2 + b2      [4000x2048]x[2048x64]
  {
    int tiles = (BN_ / 16) * (D_ / 32);
    k_gemm_bf16<<<cdiv(tiles, WPB), TPB, 0, stream>>>(
        act, FF_, W2_t, FF_, b2, 0, nullptr, ffout, D_, BN_, D_, FF_, 0);
  }
  // out = LN2(z1 + ffout)  -> [B, N, D]
  k_ln<<<cdiv(BN_, TPB), TPB, 0, stream>>>(z1, ffout, ln2_g, ln2_b, out, BN_);
}
